// CPCGNN_50766513439398
// MI455X (gfx1250) — matrix operand order repaired and checked by reading, not compiled, not run on hardware
//
#include <hip/hip_runtime.h>
#include <hip/hip_bf16.h>
#include <stdint.h>

// ---------------------------------------------------------------------------
// CDNA5 / gfx1250 WMMA types
// ---------------------------------------------------------------------------
typedef __attribute__((ext_vector_type(16))) __bf16        v16bf;
typedef __attribute__((ext_vector_type(8)))  float         v8f;
typedef __attribute__((ext_vector_type(4)))  unsigned int  u32x4;

#define GLOBAL_AS __attribute__((address_space(1)))

union BFrag {
    v16bf v;
    u32x4 q[2];
};

__device__ __forceinline__ unsigned short f2bf(float x) {
    unsigned int u = __float_as_uint(x);
    return (unsigned short)((u + 0x7FFFu + ((u >> 16) & 1u)) >> 16);  // RNE
}
__device__ __forceinline__ float sigm(float x) { return 1.0f / (1.0f + __expf(-x)); }

// Problem constants
#define BB   128
#define SS   128
#define AA   16
#define FF   16
#define HH   256
#define GG   256
#define TT   12
#define LL   16
#define NHEAD 8
#define KTOT 288     // 32 (padded x) + 256 (h)
#define NSEQ (BB*AA) // 2048
#define MBLK 16      // sequences per LSTM block (1 M-tile)

// ---------------------------------------------------------------------------
// Prep: pack LSTM weights as bf16 [1024][288] (rows: W_ih | zero-pad | W_hh)
// and combined bias. Row-per-N / contiguous-in-K == perfect B fragment.
// ---------------------------------------------------------------------------
__global__ __launch_bounds__(256) void prep_lstm_weights(
    const float* __restrict__ W_ih, const float* __restrict__ W_hh,
    const float* __restrict__ b_ih, const float* __restrict__ b_hh,
    unsigned short* __restrict__ WB, float* __restrict__ bcomb)
{
    int n = blockIdx.x * blockDim.x + threadIdx.x;
    if (n >= 4 * HH) return;
    unsigned short* row = WB + n * KTOT;
    for (int f = 0; f < FF; ++f)  row[f]      = f2bf(W_ih[n * FF + f]);
    for (int f = FF; f < 32; ++f) row[f]      = 0;
    for (int j = 0; j < HH; ++j)  row[32 + j] = f2bf(W_hh[n * HH + j]);
    bcomb[n] = b_ih[n] + b_hh[n];
}

__global__ __launch_bounds__(256) void f32_to_bf16_kernel(
    const float* __restrict__ src, unsigned short* __restrict__ dst, int n)
{
    int i = blockIdx.x * blockDim.x + threadIdx.x;
    if (i < n) dst[i] = f2bf(src[i]);
}

// ---------------------------------------------------------------------------
// LSTM: 128 blocks x 256 threads. Block owns 16 sequences (one M-tile); wave w
// owns hidden columns [16w,16w+16) and [16w+128,16w+144) of all four gates, so
// the elementwise c/h update is wave-local (c lives in VGPRs for all 128
// steps).
//
// Weight streaming: the fragment addresses are t-invariant, so LLVM would
// hoist all 72 global fragment loads out of the time loop and spill them to
// scratch (round 2), and volatile loads lower to serialized SCOPE_SYS flat
// loads (round 3). Instead we launder the WB base address through an empty
// asm block each iteration -- the address becomes opaque/loop-variant, LICM
// cannot hoist the loads, and they remain ordinary loads that lower to
// clause-batched global_load_b128 with partial s_wait_loadcnt waits, hitting
// the 576 KB L2-resident weight image every step.
// ---------------------------------------------------------------------------
__global__ __launch_bounds__(256) void lstm_kernel(
    const float* __restrict__ obs,            // (B,S,A,F)
    const unsigned short* __restrict__ WB,    // [1024][288] bf16
    const float* __restrict__ bcomb,          // [1024]
    float* __restrict__ hn,                   // [2048][256]
    unsigned short* __restrict__ hnbf)        // [2048][256] bf16
{
    __shared__ unsigned short Ast[MBLK][KTOT];  // 9 KB: [x_pad(32) | h(256)] bf16
    const int tid  = threadIdx.x;
    const int lane = tid & 31;
    const int wav  = tid >> 5;
    const int seq0 = blockIdx.x * MBLK;

    for (int idx = tid; idx < MBLK * KTOT; idx += 256)
        ((unsigned short*)Ast)[idx] = 0;      // zero pad + initial h

    const int cb0 = 16 * wav;                 // hidden col tile 0
    const int cb1 = 16 * wav + 128;           // hidden col tile 1
    const int acol = lane & 15;

    float bias[4][2];
#pragma unroll
    for (int q = 0; q < 4; ++q) {
        bias[q][0] = bcomb[q * HH + cb0 + acol];
        bias[q][1] = bcomb[q * HH + cb1 + acol];
    }

    float creg[2][8];
#pragma unroll
    for (int ct = 0; ct < 2; ++ct)
#pragma unroll
        for (int r = 0; r < 8; ++r) creg[ct][r] = 0.0f;

    const int koff  = (lane >= 16) ? 8  : 0;  // A-frag K sub-offset (ISA layout)
    const int kboff = (lane >= 16) ? 8  : 0;  // B-frag K sub-offset

    for (int t = 0; t < SS; ++t) {
        // stage x_t (16 seq x 16 feat) as bf16 -- one element per thread
        {
            int r = tid >> 4, f = tid & 15;
            int sb = seq0 + r;
            int b = sb >> 4, a = sb & 15;
            Ast[r][f] = f2bf(obs[(((size_t)b * SS + t) * AA + a) * FF + f]);
        }
        __syncthreads();                      // x + previous h visible

        // Launder the weight base address: opaque per-iteration value defeats
        // LICM (no hoist-to-scratch), loads stay plain global_load_b128.
        unsigned long long wba = (unsigned long long)WB;
        asm volatile("" : "+v"(wba));
        const GLOBAL_AS unsigned short* WBt = (const GLOBAL_AS unsigned short*)wba;

        v8f acc[4][2];
#pragma unroll
        for (int q = 0; q < 4; ++q)
#pragma unroll
            for (int ct = 0; ct < 2; ++ct)
#pragma unroll
                for (int e = 0; e < 8; ++e) acc[q][ct][e] = 0.0f;

#pragma unroll
        for (int kc = 0; kc < KTOT / 32; ++kc) {
            BFrag af;
            {
                const unsigned short* ap = &Ast[acol][kc * 32 + koff];
                af.q[0] = *reinterpret_cast<const u32x4*>(ap);
                af.q[1] = *reinterpret_cast<const u32x4*>(ap + 16);
            }
#pragma unroll
            for (int q = 0; q < 4; ++q) {
#pragma unroll
                for (int ct = 0; ct < 2; ++ct) {
                    const int ncol = q * HH + (ct ? cb1 : cb0) + acol;
                    const GLOBAL_AS unsigned short* bp =
                        WBt + (size_t)ncol * KTOT + kc * 32 + kboff;
                    BFrag bf;
                    bf.q[0] = *reinterpret_cast<const GLOBAL_AS u32x4*>(bp);
                    bf.q[1] = *reinterpret_cast<const GLOBAL_AS u32x4*>(bp + 16);
                    acc[q][ct] = __builtin_amdgcn_wmma_f32_16x16x32_bf16(
                        false, af.v, false, bf.v, (short)0, acc[q][ct], false, false);
                }
            }
        }
        __syncthreads();                      // all reads of Ast done before h writes

        // gate math + h write (C-layout: vgpr r, lane l -> row r + 8*(l/16), col l%16)
#pragma unroll
        for (int ct = 0; ct < 2; ++ct) {
            const int hid = (ct ? cb1 : cb0) + acol;
            const int rowadd = (lane >> 4) << 3;
#pragma unroll
            for (int r = 0; r < 8; ++r) {
                float zi = acc[0][ct][r] + bias[0][ct];
                float zf = acc[1][ct][r] + bias[1][ct];
                float zg = acc[2][ct][r] + bias[2][ct];
                float zo = acc[3][ct][r] + bias[3][ct];
                float cn = sigm(zf) * creg[ct][r] + sigm(zi) * tanhf(zg);
                float hv = sigm(zo) * tanhf(cn);
                creg[ct][r] = cn;
                int row = rowadd + r;
                Ast[row][32 + hid] = f2bf(hv);
                if (t == SS - 1) {
                    size_t g = (size_t)(seq0 + row) * HH + hid;
                    hn[g]   = hv;
                    hnbf[g] = f2bf(hv);
                }
            }
        }
        // next-iteration __syncthreads() makes h writes visible before reads
    }
}

// ---------------------------------------------------------------------------
// Generic bf16 WMMA GEMM: C[M][N] = A[M][K] * W[N][K]^T + bias[N]
// One 32x32 macro-tile (2x2 WMMA tiles) per wave, 8 waves per block.
// ---------------------------------------------------------------------------
__global__ __launch_bounds__(256) void gemm_bf16_kernel(
    const unsigned short* __restrict__ Abf,   // [M][K] bf16
    const unsigned short* __restrict__ Wbf,   // [N][K] bf16
    const float* __restrict__ bias,           // [N]
    float* __restrict__ C,                    // [M][N]
    int M, int N, int K)
{
    const int lane = threadIdx.x & 31;
    const int wav  = threadIdx.x >> 5;
    const int nunits = N >> 5;                // 32-wide macro-tiles in N
    const int unit = blockIdx.x * 8 + wav;
    if (unit >= (M >> 5) * nunits) return;
    const int um = unit / nunits, un = unit % nunits;

    const int koff  = (lane >= 16) ? 8  : 0;
    const int kboff = (lane >= 16) ? 8  : 0;
    const int acol  = lane & 15;

    v8f a00, a01, a10, a11;
#pragma unroll
    for (int e = 0; e < 8; ++e) { a00[e] = 0.0f; a01[e] = 0.0f; a10[e] = 0.0f; a11[e] = 0.0f; }

    const unsigned short* arow0 = Abf + (size_t)(um * 32 + acol) * K;
    const unsigned short* arow1 = arow0 + (size_t)16 * K;
    const unsigned short* brow0 = Wbf + (size_t)(un * 32 + acol) * K;
    const unsigned short* brow1 = brow0 + (size_t)16 * K;

    for (int kc = 0; kc < (K >> 5); ++kc) {
        BFrag af0, af1, bf0, bf1;
        const unsigned short* ap0 = arow0 + kc * 32 + koff;
        af0.q[0] = *reinterpret_cast<const u32x4*>(ap0);
        af0.q[1] = *reinterpret_cast<const u32x4*>(ap0 + 16);
        const unsigned short* ap1 = arow1 + kc * 32 + koff;
        af1.q[0] = *reinterpret_cast<const u32x4*>(ap1);
        af1.q[1] = *reinterpret_cast<const u32x4*>(ap1 + 16);
        const unsigned short* bp0 = brow0 + kc * 32 + kboff;
        bf0.q[0] = *reinterpret_cast<const u32x4*>(bp0);
        bf0.q[1] = *reinterpret_cast<const u32x4*>(bp0 + 16);
        const unsigned short* bp1 = brow1 + kc * 32 + kboff;
        bf1.q[0] = *reinterpret_cast<const u32x4*>(bp1);
        bf1.q[1] = *reinterpret_cast<const u32x4*>(bp1 + 16);
        a00 = __builtin_amdgcn_wmma_f32_16x16x32_bf16(false, af0.v, false, bf0.v, (short)0, a00, false, false);
        a01 = __builtin_amdgcn_wmma_f32_16x16x32_bf16(false, af0.v, false, bf1.v, (short)0, a01, false, false);
        a10 = __builtin_amdgcn_wmma_f32_16x16x32_bf16(false, af1.v, false, bf0.v, (short)0, a10, false, false);
        a11 = __builtin_amdgcn_wmma_f32_16x16x32_bf16(false, af1.v, false, bf1.v, (short)0, a11, false, false);
    }
    const int col0 = un * 32 + acol;
    const int col1 = col0 + 16;
    const float bv0 = bias[col0];
    const float bv1 = bias[col1];
    const int rb0 = um * 32 + ((lane >> 4) << 3);
    const int rb1 = rb0 + 16;
#pragma unroll
    for (int r = 0; r < 8; ++r) {
        C[(size_t)(rb0 + r) * N + col0] = a00[r] + bv0;
        C[(size_t)(rb0 + r) * N + col1] = a01[r] + bv1;
        C[(size_t)(rb1 + r) * N + col0] = a10[r] + bv0;
        C[(size_t)(rb1 + r) * N + col1] = a11[r] + bv1;
    }
}

// ---------------------------------------------------------------------------
// GATv2 layer-1 attention (8 heads). One block per (b,i); wave = head.
// ---------------------------------------------------------------------------
__global__ __launch_bounds__(256) void gat1_attn_kernel(
    const float* __restrict__ src1,   // [2048][2048]  (b*16+j, h*256+f)
    const float* __restrict__ dst1,   // [2048][2048]
    const float* __restrict__ attn1,  // [8][256]
    float* __restrict__ r1,           // [2048][256]
    unsigned short* __restrict__ r1bf)
{
    __shared__ float s_sc[NHEAD][16];
    __shared__ float s_out[NHEAD][HH];
    const int bi = blockIdx.x;
    const int b = bi >> 4, i = bi & 15;
    const int lane = threadIdx.x & 31, hh = threadIdx.x >> 5;

    float dstv[8], attv[8];
#pragma unroll
    for (int k = 0; k < 8; ++k) {
        int f = lane + 32 * k;
        dstv[k] = dst1[(size_t)bi * 2048 + hh * HH + f];
        attv[k] = attn1[hh * HH + f];
    }
    for (int j = 0; j < 16; ++j) {
        float p = 0.0f;
        const float* sp = src1 + (size_t)(b * 16 + j) * 2048 + hh * HH;
#pragma unroll
        for (int k = 0; k < 8; ++k) {
            float e = dstv[k] + sp[lane + 32 * k];
            p += attv[k] * (e > 0.0f ? e : 0.2f * e);
        }
        for (int off = 16; off; off >>= 1) p += __shfl_xor(p, off);
        if (lane == 0) s_sc[hh][j] = (j == i) ? -1e9f : p;
    }
    __syncthreads();
    float mx = -1e30f;
    for (int j = 0; j < 16; ++j) mx = fmaxf(mx, s_sc[hh][j]);
    float al[16], ssum = 0.0f;
    for (int j = 0; j < 16; ++j) { al[j] = __expf(s_sc[hh][j] - mx); ssum += al[j]; }
    float inv = 1.0f / ssum;
#pragma unroll
    for (int k = 0; k < 8; ++k) {
        int f = lane + 32 * k;
        float a = 0.0f;
        for (int j = 0; j < 16; ++j)
            a += al[j] * src1[(size_t)(b * 16 + j) * 2048 + hh * HH + f];
        s_out[hh][f] = a * inv;
    }
    __syncthreads();
    int f = threadIdx.x;
    float m = 0.0f;
#pragma unroll
    for (int h = 0; h < NHEAD; ++h) m += s_out[h][f];
    m *= (1.0f / NHEAD);
    r1[(size_t)bi * HH + f]   = m;
    r1bf[(size_t)bi * HH + f] = f2bf(m);
}

// ---------------------------------------------------------------------------
// GATv2 layer-2 attention (1 head). One block per (b,i); wave w does j=2w,2w+1.
// ---------------------------------------------------------------------------
__global__ __launch_bounds__(256) void gat2_attn_kernel(
    const float* __restrict__ src2,   // [2048][256]
    const float* __restrict__ dst2,
    const float* __restrict__ attn2,  // [256]
    float* __restrict__ r2)           // [2048][256]
{
    __shared__ float s_sc[16];
    const int bi = blockIdx.x;
    const int b = bi >> 4, i = bi & 15;
    const int lane = threadIdx.x & 31, wav = threadIdx.x >> 5;

    float dstv[8], attv[8];
#pragma unroll
    for (int k = 0; k < 8; ++k) {
        int f = lane + 32 * k;
        dstv[k] = dst2[(size_t)bi * GG + f];
        attv[k] = attn2[f];
    }
#pragma unroll
    for (int jj = 0; jj < 2; ++jj) {
        int j = wav * 2 + jj;
        float p = 0.0f;
        const float* sp = src2 + (size_t)(b * 16 + j) * GG;
#pragma unroll
        for (int k = 0; k < 8; ++k) {
            float e = dstv[k] + sp[lane + 32 * k];
            p += attv[k] * (e > 0.0f ? e : 0.2f * e);
        }
        for (int off = 16; off; off >>= 1) p += __shfl_xor(p, off);
        if (lane == 0) s_sc[j] = (j == i) ? -1e9f : p;
    }
    __syncthreads();
    float mx = -1e30f;
    for (int j = 0; j < 16; ++j) mx = fmaxf(mx, s_sc[j]);
    float al[16], ssum = 0.0f;
    for (int j = 0; j < 16; ++j) { al[j] = __expf(s_sc[j] - mx); ssum += al[j]; }
    float inv = 1.0f / ssum;
    int f = threadIdx.x;
    float a = 0.0f;
    for (int j = 0; j < 16; ++j)
        a += al[j] * src2[(size_t)(b * 16 + j) * GG + f];
    r2[(size_t)bi * GG + f] = a * inv;
}

// ---------------------------------------------------------------------------
// Node mean-pool -> agent_context; write concat output; zero loss slot.
// ---------------------------------------------------------------------------
__global__ __launch_bounds__(256) void ctx_kernel(
    const float* __restrict__ r2,        // [2048][256]
    const float* __restrict__ hideout,   // [128][2]
    const float* __restrict__ timestep,  // [128][1]
    float* __restrict__ ctx,             // [128][256]
    float* __restrict__ out)             // [128*259 + 1]
{
    const int b = blockIdx.x;
    const int g = threadIdx.x;
    float s = 0.0f;
    for (int i = 0; i < 16; ++i) s += r2[(size_t)(b * 16 + i) * GG + g];
    s *= (1.0f / 16.0f);
    ctx[b * GG + g]   = s;
    out[b * 259 + g]  = s;
    if (g == 0) {
        out[b * 259 + 256] = hideout[b * 2];
        out[b * 259 + 257] = hideout[b * 2 + 1];
        out[b * 259 + 258] = timestep[b];
        if (b == 0) out[BB * 259] = 0.0f;
    }
}

// ---------------------------------------------------------------------------
// CPC InfoNCE loss. One block per t.
// ---------------------------------------------------------------------------
__global__ __launch_bounds__(256) void cpc_kernel(
    const float* __restrict__ ctx,     // [128][256]
    const float* __restrict__ future,  // [128][12][2]
    const float* __restrict__ Wk,      // [12][16][256]
    const float* __restrict__ bk,      // [12][16]
    const float* __restrict__ Wl,      // [16][2]
    const float* __restrict__ bl,      // [16]
    float* __restrict__ loss_out)
{
    __shared__ float pn[BB][LL];   // normalized pred
    __shared__ float tn[BB][LL];   // normalized proj
    __shared__ float red[256];
    const int t = blockIdx.x;
    const int tid = threadIdx.x;

    for (int idx = tid; idx < BB * LL; idx += 256) {
        int bb = idx >> 4, l = idx & 15;
        const float* wrow = Wk + (size_t)(t * LL + l) * GG;
        const float* crow = ctx + (size_t)bb * GG;
        float acc = bk[t * LL + l];
        for (int g = 0; g < GG; ++g) acc += wrow[g] * crow[g];
        pn[bb][l] = acc;
        float f0 = future[(bb * TT + t) * 2];
        float f1 = future[(bb * TT + t) * 2 + 1];
        tn[bb][l] = f0 * Wl[l * 2] + f1 * Wl[l * 2 + 1] + bl[l];
    }
    __syncthreads();
    for (int bb = tid; bb < BB; bb += 256) {
        float s = 0.0f;
        for (int l = 0; l < LL; ++l) s += pn[bb][l] * pn[bb][l];
        float inv = 1.0f / fmaxf(sqrtf(s), 1e-12f);
        for (int l = 0; l < LL; ++l) pn[bb][l] *= inv;
        s = 0.0f;
        for (int l = 0; l < LL; ++l) s += tn[bb][l] * tn[bb][l];
        inv = 1.0f / fmaxf(sqrtf(s), 1e-12f);
        for (int l = 0; l < LL; ++l) tn[bb][l] *= inv;
    }
    __syncthreads();
    float part = 0.0f;
    for (int bb = tid; bb < BB; bb += 256) {
        float mx = -1e30f, diag = 0.0f;
        for (int c = 0; c < BB; ++c) {
            float d = 0.0f;
            for (int l = 0; l < LL; ++l) d += tn[bb][l] * pn[c][l];
            mx = fmaxf(mx, d);
            if (c == bb) diag = d;
        }
        float se = 0.0f;
        for (int c = 0; c < BB; ++c) {
            float d = 0.0f;
            for (int l = 0; l < LL; ++l) d += tn[bb][l] * pn[c][l];
            se += __expf(d - mx);
        }
        part += diag - mx - __logf(se);
    }
    red[tid] = part;
    __syncthreads();
    for (int s = 128; s; s >>= 1) {
        if (tid < s) red[tid] += red[tid + s];
        __syncthreads();
    }
    if (tid == 0) atomicAdd(loss_out, -red[0] / (float)(BB * TT));
}

// ---------------------------------------------------------------------------
// Launch
// ---------------------------------------------------------------------------
extern "C" void kernel_launch(void* const* d_in, const int* in_sizes, int n_in,
                              void* d_out, int out_size, void* d_ws, size_t ws_size,
                              hipStream_t stream) {
    (void)in_sizes; (void)n_in; (void)out_size; (void)ws_size;
    const float* agent_obs = (const float*)d_in[0];
    const float* future    = (const float*)d_in[1];
    const float* hideout   = (const float*)d_in[2];
    const float* timestep  = (const float*)d_in[3];
    // d_in[4] = num_agents (16, constant)
    const float* W_ih  = (const float*)d_in[5];
    const float* W_hh  = (const float*)d_in[6];
    const float* b_ih  = (const float*)d_in[7];
    const float* b_hh  = (const float*)d_in[8];
    const float* Wsrc1 = (const float*)d_in[9];
    const float* bsrc1 = (const float*)d_in[10];
    const float* Wdst1 = (const float*)d_in[11];
    const float* bdst1 = (const float*)d_in[12];
    const float* attn1 = (const float*)d_in[13];
    const float* Wsrc2 = (const float*)d_in[14];
    const float* bsrc2 = (const float*)d_in[15];
    const float* Wdst2 = (const float*)d_in[16];
    const float* bdst2 = (const float*)d_in[17];
    const float* attn2 = (const float*)d_in[18];
    const float* Wk    = (const float*)d_in[19];
    const float* bk    = (const float*)d_in[20];
    const float* Wl    = (const float*)d_in[21];
    const float* bl    = (const float*)d_in[22];
    float* out = (float*)d_out;

    char* ws = (char*)d_ws;
    size_t off = 0;
    auto take = [&](size_t bytes) -> char* {
        char* p = ws + off;
        off = (off + bytes + 255) & ~(size_t)255;
        return p;
    };
    unsigned short* WB    = (unsigned short*)take((size_t)1024 * KTOT * 2);
    float*          bcomb = (float*)take(1024 * 4);
    unsigned short* Ws1b  = (unsigned short*)take((size_t)2048 * 256 * 2);
    unsigned short* Wd1b  = (unsigned short*)take((size_t)2048 * 256 * 2);
    unsigned short* Ws2b  = (unsigned short*)take((size_t)256 * 256 * 2);
    unsigned short* Wd2b  = (unsigned short*)take((size_t)256 * 256 * 2);
    float*          hn    = (float*)take((size_t)NSEQ * HH * 4);
    unsigned short* hnbf  = (unsigned short*)take((size_t)NSEQ * HH * 2);
    float*          src1  = (float*)take((size_t)NSEQ * 2048 * 4);
    float*          dst1  = (float*)take((size_t)NSEQ * 2048 * 4);
    float*          r1    = (float*)take((size_t)NSEQ * HH * 4);
    unsigned short* r1bf  = (unsigned short*)take((size_t)NSEQ * HH * 2);
    float*          src2  = (float*)take((size_t)NSEQ * GG * 4);
    float*          dst2  = (float*)take((size_t)NSEQ * GG * 4);
    float*          r2    = (float*)take((size_t)NSEQ * GG * 4);
    float*          ctx   = (float*)take((size_t)BB * GG * 4);

    prep_lstm_weights<<<4, 256, 0, stream>>>(W_ih, W_hh, b_ih, b_hh, WB, bcomb);
    f32_to_bf16_kernel<<<(2048 * 256 + 255) / 256, 256, 0, stream>>>(Wsrc1, Ws1b, 2048 * 256);
    f32_to_bf16_kernel<<<(2048 * 256 + 255) / 256, 256, 0, stream>>>(Wdst1, Wd1b, 2048 * 256);
    f32_to_bf16_kernel<<<(256 * 256 + 255) / 256, 256, 0, stream>>>(Wsrc2, Ws2b, 256 * 256);
    f32_to_bf16_kernel<<<(256 * 256 + 255) / 256, 256, 0, stream>>>(Wdst2, Wd2b, 256 * 256);

    lstm_kernel<<<NSEQ / MBLK, 256, 0, stream>>>(agent_obs, WB, bcomb, hn, hnbf);

    // layer-1 GEMMs: 2048x2048x256, 32x32 macro-tile per wave
    gemm_bf16_kernel<<<(2048 / 32) * (2048 / 32) / 8, 256, 0, stream>>>(hnbf, Ws1b, bsrc1, src1, 2048, 2048, 256);
    gemm_bf16_kernel<<<(2048 / 32) * (2048 / 32) / 8, 256, 0, stream>>>(hnbf, Wd1b, bdst1, dst1, 2048, 2048, 256);
    gat1_attn_kernel<<<NSEQ, 256, 0, stream>>>(src1, dst1, attn1, r1, r1bf);

    // layer-2 GEMMs: 2048x256x256
    gemm_bf16_kernel<<<(2048 / 32) * (256 / 32) / 8, 256, 0, stream>>>(r1bf, Ws2b, bsrc2, src2, 2048, 256, 256);
    gemm_bf16_kernel<<<(2048 / 32) * (256 / 32) / 8, 256, 0, stream>>>(r1bf, Wd2b, bdst2, dst2, 2048, 256, 256);
    gat2_attn_kernel<<<NSEQ, 256, 0, stream>>>(src2, dst2, attn2, r2);

    ctx_kernel<<<BB, 256, 0, stream>>>(r2, hideout, timestep, ctx, out);
    cpc_kernel<<<TT, 256, 0, stream>>>(ctx, future, Wk, bk, Wl, bl, out + BB * 259);
}
